// qRIMBlock_19318762897711
// MI455X (gfx1250) — matrix operand order repaired and checked by reading, not compiled, not run on hardware
//
#include <hip/hip_runtime.h>
#include <math.h>

typedef _Float16 h16;
typedef __attribute__((ext_vector_type(16))) _Float16 v16h;
typedef __attribute__((ext_vector_type(8)))  float    v8f;

#define XD 256
#define YD 256
#define BD 2
#define ED 4
#define CDN 12
#define FD 64
#define NPIX (XD*YD)

// ---------------- WMMA fragment helpers (CDNA5 16x16x32 f16 layouts) ----------------
// A: 16x32 f16, row-major, stride lda (literal at call sites -> vectorizes to b128 pairs).
__device__ __forceinline__ v16h frag_a(const h16* __restrict__ A, int lda, int lane) {
  int row = lane & 15, hk = (lane >> 4) & 1;
  v16h a;
#pragma unroll
  for (int i = 0; i < 8; ++i) {
    int kb = hk * 16 + i * 2;
    a[2*i]   = A[row*lda + kb];
    a[2*i+1] = A[row*lda + kb + 1];
  }
  return a;
}
// B tile stored TRANSPOSED in LDS: Bm[n*ldk + k], ldk multiple of 8 (16B-aligned rows).
// Lane reads 16 contiguous halves -> two ds_load_b128.
__device__ __forceinline__ v16h frag_bt(const h16* __restrict__ Bm, int ldk, int ks, int lane) {
  int col = lane & 15, ko = ks*32 + (lane >> 4) * 16;
  const h16* p = Bm + col*ldk + ko;
  v16h b;
#pragma unroll
  for (int h = 0; h < 16; ++h) b[h] = p[h];
  return b;
}
__device__ __forceinline__ v8f wmma16(v16h a, v16h b, v8f c) {
  return __builtin_amdgcn_wmma_f32_16x16x32_f16(false, a, false, b, (short)0, c, false, false);
}

// ---------------- 256-point radix-2 FFT in LDS, 128 threads ----------------
__device__ inline void fft256(float2* s, int tid, int inverse) {
  __syncthreads();
  for (int i = tid; i < 256; i += 128) {
    int j = (int)(__brev((unsigned)i) >> 24);
    if (j > i) { float2 t = s[i]; s[i] = s[j]; s[j] = t; }
  }
  __syncthreads();
#pragma unroll
  for (int st = 0; st < 8; ++st) {
    int hstep = 1 << st;
    int pos = tid & (hstep - 1);
    int i0 = ((tid >> st) << (st + 1)) + pos;
    int i1 = i0 + hstep;
    float ang = (inverse ? 3.14159265358979f : -3.14159265358979f) * (float)pos / (float)hstep;
    float sw, cw; __sincosf(ang, &sw, &cw);
    float2 a = s[i0], b = s[i1];
    float2 t = make_float2(b.x*cw - b.y*sw, b.x*sw + b.y*cw);
    s[i1] = make_float2(a.x - t.x, a.y - t.y);
    s[i0] = make_float2(a.x + t.x, a.y + t.y);
    __syncthreads();
  }
}

// ---------------- weight conversion (fp32 -> f16, repacked) ----------------
__global__ void k_convert_weights(const float* W1, const float* Wih1, const float* Whh1,
                                  const float* W2, const float* Wih2, const float* Whh2,
                                  const float* Wf, h16* out) {
  int idx = blockIdx.x * blockDim.x + threadIdx.x;
  const int n_w1 = 64*224, n_g = 192*64, n_w2 = 9*64*64, n_wf = 9*16*64;
  if (idx < n_w1) {                       // W1 [64][224], k = c*25 + tap, pad 200..223
    int m = idx / 224, k = idx % 224;
    float v = 0.f;
    if (k < 200) { int c = k / 25, t = k % 25; v = W1[(m*8 + c)*25 + t]; }
    out[idx] = (h16)v; return;
  }
  idx -= n_w1;
  if (idx < n_g) { out[n_w1 + idx] = (h16)Wih1[idx]; return; }
  idx -= n_g;
  if (idx < n_g) { out[n_w1 + n_g + idx] = (h16)Whh1[idx]; return; }
  idx -= n_g;
  if (idx < n_g) { out[n_w1 + 2*n_g + idx] = (h16)Wih2[idx]; return; }
  idx -= n_g;
  if (idx < n_g) { out[n_w1 + 3*n_g + idx] = (h16)Whh2[idx]; return; }
  idx -= n_g;
  if (idx < n_w2) {                       // W2 -> [tap][64][64]
    int tap = idx / 4096, r = idx % 4096, m = r / 64, c = r % 64;
    out[n_w1 + 4*n_g + idx] = (h16)W2[(m*64 + c)*9 + tap]; return;
  }
  idx -= n_w2;
  if (idx < n_wf) {                       // Wf -> [tap][16(pad)][64]
    int tap = idx / 1024, r = idx % 1024, m = r / 64, c = r % 64;
    float v = (m < 4) ? Wf[(m*64 + c)*9 + tap] : 0.f;
    out[n_w1 + 4*n_g + n_w2 + idx] = (h16)v;
  }
}

// ---------------- state init ----------------
__global__ void k_init(float* h1, float* h2, float* pred,
                       const float* R2, const float* S0, const float* B0, const float* phi) {
  int idx = blockIdx.x * blockDim.x + threadIdx.x;
  if (idx < BD*FD*NPIX) { h1[idx] = 0.f; h2[idx] = 0.f; }
  if (idx < BD*4*NPIX) {
    int y = idx & 255, x = (idx >> 8) & 255, ch = (idx >> 16) & 3, b = idx >> 18;
    int mi = (b*XD + x)*YD + y;
    pred[idx] = (ch==0) ? R2[mi] : (ch==1) ? S0[mi] : (ch==2) ? B0[mi] : phi[mi];
  }
}

// ---------------- MEGRE signal model ----------------
__global__ void k_sigmodel(const float* R2m, const float* S0m, const float* B0m, const float* phim,
                           const float* TEs, const float* gamma, float2* ps, float2* base) {
  int idx = blockIdx.x*blockDim.x + threadIdx.x;
  if (idx >= BD*ED*NPIX) return;
  int y = idx & 255, x = (idx >> 8) & 255, e = (idx >> 16) & 3, b = idx >> 18;
  int mi = (b*XD + x)*YD + y;
  float te = TEs[e] * 1e-3f;
  float R2 = R2m[mi]*gamma[0], S0 = S0m[mi]*gamma[1], B0v = B0m[mi]*gamma[2], ph = phim[mi]*gamma[3];
  float d = __expf(-te * R2);
  float sw, cw; __sincosf(-te * B0v, &sw, &cw);
  float2 bs = make_float2(d*cw, d*sw);
  base[idx] = bs;
  ps[idx] = make_float2(S0*bs.x - ph*bs.y, S0*bs.y + ph*bs.x);
}

// ---------------- FFT passes ----------------
__global__ void k_fft_row_fwd(const float2* ps, const float* sens, float2* buf) {
  __shared__ float2 s[256];
  int tid = threadIdx.x, t = blockIdx.x;
  int x = t & 255; t >>= 8;
  int c = t % CDN; t /= CDN;
  int e = t & 3, b = t >> 2;
  int psrow = ((b*ED + e)*XD + x)*YD;
  int srow  = (((b*CDN + c)*XD + x)*YD) * 2;
  for (int y = tid; y < 256; y += 128) {
    float2 p = ps[psrow + y];
    float sr = sens[srow + y*2], si = sens[srow + y*2 + 1];
    s[y] = make_float2(p.x*sr - p.y*si, p.x*si + p.y*sr);
  }
  fft256(s, tid, 0);
  float2* dst = buf + (size_t)blockIdx.x * 256;
  for (int y = tid; y < 256; y += 128) dst[y] = s[y];
}

__global__ void k_fft_col_resid(float2* buf, const float* ksp, const float* mask) {
  __shared__ float2 s[256];
  int tid = threadIdx.x, t = blockIdx.x;
  int y = t & 255; t >>= 8;
  int c = t % CDN; t /= CDN;
  int e = t & 3, b = t >> 2;
  size_t img = (size_t)((b*ED + e)*CDN + c) * NPIX;
  for (int x = tid; x < 256; x += 128) s[x] = buf[img + (size_t)x*256 + y];
  fft256(s, tid, 0);
  for (int x = tid; x < 256; x += 128) {
    size_t ki = (img + (size_t)x*256 + y) * 2;
    float m = mask[(b*XD + x)*YD + y];
    float2 k = s[x];
    s[x] = make_float2((k.x - ksp[ki]) * m, (k.y - ksp[ki+1]) * m);
  }
  fft256(s, tid, 1);
  const float sc = 1.0f/256.0f;
  for (int x = tid; x < 256; x += 128) {
    float2 v = s[x];
    buf[img + (size_t)x*256 + y] = make_float2(v.x*sc, v.y*sc);
  }
}

__global__ void k_ifft_row_cc(const float2* buf, const float* sens, float2* cc) {
  __shared__ float2 s[256];
  int tid = threadIdx.x, t = blockIdx.x;
  int x = t & 255; t >>= 8;
  int e = t & 3, b = t >> 2;
  float2 acc0 = make_float2(0.f,0.f), acc1 = make_float2(0.f,0.f);
  for (int c = 0; c < CDN; ++c) {
    __syncthreads();
    const float2* src = buf + ((size_t)((b*ED + e)*CDN + c) * NPIX + (size_t)x*256);
    for (int y = tid; y < 256; y += 128) s[y] = src[y];
    fft256(s, tid, 1);
    const float sc = 1.0f/256.0f;
    int srow = (((b*CDN + c)*XD + x)*YD)*2;
#pragma unroll
    for (int rep = 0; rep < 2; ++rep) {
      int y = tid + rep*128;
      float2 v = s[y]; v.x *= sc; v.y *= sc;
      float sr = sens[srow + y*2], si = sens[srow + y*2+1];
      float2 r = make_float2(v.x*sr + v.y*si, v.y*sr - v.x*si);   // v * conj(sens)
      if (rep == 0) { acc0.x += r.x; acc0.y += r.y; } else { acc1.x += r.x; acc1.y += r.y; }
    }
  }
  size_t orow = ((size_t)(b*ED + e)*XD + x)*YD;
  cc[orow + tid] = acc0;
  cc[orow + tid + 128] = acc1;
}

__global__ void k_grad(const float2* cc, const float2* ps, const float2* base,
                       const float* TEs, float* grad) {
  int idx = blockIdx.x*blockDim.x + threadIdx.x;
  if (idx >= BD*NPIX) return;
  int y = idx & 255, x = (idx >> 8) & 255, b = idx >> 16;
  float gR2=0.f, gS0=0.f, gB0=0.f, gph=0.f;
  for (int e = 0; e < ED; ++e) {
    size_t i = ((size_t)(b*ED + e)*XD + x)*YD + y;
    float te = TEs[e]*1e-3f;
    float2 p = ps[i], bs = base[i], c = cc[i];
    gR2 += -te * (p.x*c.x + p.y*c.y);
    gS0 += bs.x*c.x + bs.y*c.y;
    gph += bs.x*c.y - bs.y*c.x;
    gB0 += te * (p.y*c.x - p.x*c.y);
  }
  float g[4] = {gR2*0.01f, gS0*0.01f, gB0*0.01f, gph*0.01f};
  for (int ch = 0; ch < 4; ++ch) {
    float v = g[ch];
    if (!(v == v)) v = 0.f;
    grad[((size_t)(b*4 + ch)*XD + x)*YD + y] = v;
  }
}

// ---------------- conv 5x5 (8->64) + bias + relu, implicit GEMM K=224(pad), WMMA ----------------
__global__ __launch_bounds__(128) void k_conv5(const float* __restrict__ grad,
                                               const float* __restrict__ pred,
                                               float* __restrict__ out,
                                               const h16* __restrict__ W1h,
                                               const float* __restrict__ b1) {
  __shared__ __align__(16) h16 Bt[16*232];            // transposed: [n][k], ldk=232
  int tid = threadIdx.x, lane = tid & 31, wave = tid >> 5;
  int bid = blockIdx.x;
  int yt = bid & 15, x = (bid >> 4) & 255, b = bid >> 12;
  int y0 = yt * 16;
  for (int i = tid; i < 16*224; i += 128) {
    int n = i / 224, k = i % 224;
    float v = 0.f;
    if (k < 200) {
      int ci = k / 25, tp = k % 25, tx = tp / 5, ty = tp % 5;
      int xx = x + tx - 2, yy = y0 + n + ty - 2;
      if (xx >= 0 && xx < XD && yy >= 0 && yy < YD) {
        size_t ii = ((size_t)(b*4 + (ci & 3))*XD + xx)*YD + yy;
        v = (ci < 4) ? grad[ii] : pred[ii];
      }
    }
    Bt[n*232 + k] = (h16)v;
  }
  __syncthreads();
  v8f acc = {};
  int mt = wave;
#pragma unroll
  for (int ks = 0; ks < 7; ++ks) {
    v16h a  = frag_a(W1h + mt*16*224 + ks*32, 224, lane);
    v16h bf = frag_bt(Bt, 232, ks, lane);
    acc = wmma16(a, bf, acc);
  }
  int col = lane & 15, mo = (lane >> 4) * 8;
#pragma unroll
  for (int r = 0; r < 8; ++r) {
    int ch = mt*16 + mo + r;
    out[((size_t)(b*FD + ch)*XD + x)*YD + y0 + col] = fmaxf(acc[r] + b1[ch], 0.f);
  }
}

// ---------------- fused ConvGRU: two 1x1 GEMMs (192x64) + gates, WMMA ----------------
__global__ __launch_bounds__(128) void k_gru(const float* __restrict__ tin,
                                             float* __restrict__ h,
                                             const h16* __restrict__ Wih,
                                             const h16* __restrict__ Whh,
                                             const float* __restrict__ bih,
                                             const float* __restrict__ bhh) {
  __shared__ __align__(16) h16 Bt[16*72];             // transposed: [n][ch], ldk=72
  __shared__ __align__(16) h16 Bh[16*72];
  __shared__ float Hf[64*16];
  __shared__ float Gi[192*16];
  __shared__ float Gh[192*16];
  int tid = threadIdx.x, lane = tid & 31, wave = tid >> 5;
  int bid = blockIdx.x;
  int yt = bid & 15, x = (bid >> 4) & 255, b = bid >> 12;
  int y0 = yt*16;
  for (int idx = tid; idx < 64*4; idx += 128) {       // 256 float4 chunks (4 y each)
    int ch = idx >> 2, q = idx & 3;
    size_t row = ((size_t)(b*FD + ch)*XD + x)*YD + y0 + q*4;
    float4 tv = *(const float4*)(tin + row);
    float4 hv = *(const float4*)(h + row);
#pragma unroll
    for (int j = 0; j < 4; ++j) {
      int n = q*4 + j;
      float tvj = (&tv.x)[j], hvj = (&hv.x)[j];
      Bt[n*72 + ch] = (h16)tvj;
      Bh[n*72 + ch] = (h16)hvj;
      Hf[ch*16 + n] = hvj;
    }
  }
  __syncthreads();
  for (int mt = wave; mt < 12; mt += 4) {
    v8f ai = {}, ah = {};
#pragma unroll
    for (int ks = 0; ks < 2; ++ks) {
      v16h bfa = frag_bt(Bt, 72, ks, lane);
      v16h bfb = frag_bt(Bh, 72, ks, lane);
      v16h wa  = frag_a(Wih + mt*16*64 + ks*32, 64, lane);
      v16h wb  = frag_a(Whh + mt*16*64 + ks*32, 64, lane);
      ai = wmma16(wa, bfa, ai);
      ah = wmma16(wb, bfb, ah);
    }
    int col = lane & 15, mo = (lane >> 4) * 8;
#pragma unroll
    for (int r = 0; r < 8; ++r) {
      Gi[(mt*16 + mo + r)*16 + col] = ai[r];
      Gh[(mt*16 + mo + r)*16 + col] = ah[r];
    }
  }
  __syncthreads();
  for (int i = tid; i < 64*16; i += 128) {
    int ch = i >> 4, n = i & 15;
    float r  = 1.f/(1.f + __expf(-((Gi[ch*16+n]       + bih[ch])      + (Gh[ch*16+n]       + bhh[ch]))));
    float z  = 1.f/(1.f + __expf(-((Gi[(64+ch)*16+n]  + bih[64+ch])   + (Gh[(64+ch)*16+n]  + bhh[64+ch]))));
    float nn = tanhf((Gi[(128+ch)*16+n] + bih[128+ch]) + r*(Gh[(128+ch)*16+n] + bhh[128+ch]));
    float hv = Hf[i];
    h[((size_t)(b*FD + ch)*XD + x)*YD + y0 + n] = (1.f - z)*nn + z*hv;
  }
}

// ---------------- conv 3x3 dilated (64->64) + bias + relu, tap-accumulated WMMA ----------------
__global__ __launch_bounds__(128) void k_conv3(const float* __restrict__ in,
                                               float* __restrict__ out,
                                               const h16* __restrict__ W,
                                               const float* __restrict__ bias, int dil) {
  __shared__ __align__(16) h16 Bt[2][16*72];          // double-buffered transposed tiles
  int tid = threadIdx.x, lane = tid & 31, wave = tid >> 5;
  int bid = blockIdx.x;
  int yt = bid & 15, x = (bid >> 4) & 255, b = bid >> 12;
  int y0 = yt*16;
  v8f acc = {};
  int mt = wave;
  for (int tap = 0; tap < 9; ++tap) {
    int tx = tap / 3, ty = tap % 3;
    int xx = x + (tx - 1)*dil;
    h16* Bcur = Bt[tap & 1];
    // prefetch next tap's rows into cache
    if (tap + 1 < 9) {
      int tx2 = (tap+1) / 3;
      int xx2 = x + (tx2 - 1)*dil;
      if (xx2 >= 0 && xx2 < XD) {
        int ch = tid & 63;
        __builtin_prefetch(in + ((size_t)(b*FD + ch)*XD + xx2)*YD + y0, 0, 1);
      }
    }
    for (int i = tid; i < 64*16; i += 128) {
      int ch = i >> 4, n = i & 15;
      int yy = y0 + n + (ty - 1)*dil;
      float v = 0.f;
      if (xx >= 0 && xx < XD && yy >= 0 && yy < YD)
        v = in[((size_t)(b*FD + ch)*XD + xx)*YD + yy];
      Bcur[n*72 + ch] = (h16)v;
    }
    __syncthreads();
#pragma unroll
    for (int ks = 0; ks < 2; ++ks) {
      v16h a  = frag_a(W + tap*64*64 + mt*16*64 + ks*32, 64, lane);
      v16h bf = frag_bt(Bcur, 72, ks, lane);
      acc = wmma16(a, bf, acc);
    }
  }
  int col = lane & 15, mo = (lane >> 4)*8;
#pragma unroll
  for (int r = 0; r < 8; ++r) {
    int ch = mt*16 + mo + r;
    out[((size_t)(b*FD + ch)*XD + x)*YD + y0 + col] = fmaxf(acc[r] + bias[ch], 0.f);
  }
}

// ---------------- final conv 3x3 (64->4, M padded to 16) fused with pred update ----------------
__global__ __launch_bounds__(32) void k_convf_update(const float* __restrict__ h2,
                                                     const h16* __restrict__ Wfh,
                                                     float* __restrict__ pred,
                                                     float* __restrict__ outstep) {
  __shared__ __align__(16) h16 Bt[2][16*72];
  int lane = threadIdx.x;                 // 32 threads, 1 wave
  int bid = blockIdx.x;
  int yt = bid & 15, x = (bid >> 4) & 255, b = bid >> 12;
  int y0 = yt*16;
  v8f acc = {};
  for (int tap = 0; tap < 9; ++tap) {
    int tx = tap/3, ty = tap%3;
    int xx = x + tx - 1;
    h16* Bcur = Bt[tap & 1];
    for (int i = lane; i < 64*16; i += 32) {
      int ch = i >> 4, n = i & 15;
      int yy = y0 + n + ty - 1;
      float v = 0.f;
      if (xx >= 0 && xx < XD && yy >= 0 && yy < YD)
        v = h2[((size_t)(b*FD + ch)*XD + xx)*YD + yy];
      Bcur[n*72 + ch] = (h16)v;
    }
    __syncthreads();
#pragma unroll
    for (int ks = 0; ks < 2; ++ks) {
      v16h a  = frag_a(Wfh + tap*16*64 + ks*32, 64, lane);
      v16h bf = frag_bt(Bcur, 72, ks, lane);
      acc = wmma16(a, bf, acc);
    }
  }
  int col = lane & 15, mo = (lane >> 4)*8;
#pragma unroll
  for (int r = 0; r < 8; ++r) {
    int ch = mo + r;
    if (ch < 4) {
      size_t pi = ((size_t)(b*4 + ch)*XD + x)*YD + y0 + col;
      float p = pred[pi] + acc[r];
      if (ch == 0) p = fmaxf(p, 0.f);
      pred[pi] = p;
      outstep[pi] = p;
    }
  }
}

// ---------------- host launcher ----------------
extern "C" void kernel_launch(void* const* d_in, const int* in_sizes, int n_in,
                              void* d_out, int out_size, void* d_ws, size_t ws_size,
                              hipStream_t stream) {
  (void)in_sizes; (void)n_in; (void)out_size; (void)ws_size;
  const float* ksp   = (const float*)d_in[0];
  const float* R2i   = (const float*)d_in[1];
  const float* S0i   = (const float*)d_in[2];
  const float* B0i   = (const float*)d_in[3];
  const float* phii  = (const float*)d_in[4];
  const float* TEs   = (const float*)d_in[5];
  const float* sens  = (const float*)d_in[6];
  const float* mask  = (const float*)d_in[7];
  const float* gamma = (const float*)d_in[8];
  const float* W1    = (const float*)d_in[9];
  const float* b1    = (const float*)d_in[10];
  const float* Wih1  = (const float*)d_in[11];
  const float* Whh1  = (const float*)d_in[12];
  const float* bih1  = (const float*)d_in[13];
  const float* bhh1  = (const float*)d_in[14];
  const float* W2    = (const float*)d_in[15];
  const float* b2    = (const float*)d_in[16];
  const float* Wih2  = (const float*)d_in[17];
  const float* Whh2  = (const float*)d_in[18];
  const float* bih2  = (const float*)d_in[19];
  const float* bhh2  = (const float*)d_in[20];
  const float* Wf    = (const float*)d_in[21];
  float* out = (float*)d_out;

  char* ws = (char*)d_ws;
  size_t off = 0;
  auto alloc = [&](size_t bytes) { char* p = ws + off; off += (bytes + 255) & ~(size_t)255; return p; };
  float2* buf  = (float2*)alloc((size_t)BD*ED*CDN*NPIX*sizeof(float2));   // 2D-FFT scratch
  float2* ps   = (float2*)alloc((size_t)BD*ED*NPIX*sizeof(float2));
  float2* base = (float2*)alloc((size_t)BD*ED*NPIX*sizeof(float2));
  float2* cc   = (float2*)alloc((size_t)BD*ED*NPIX*sizeof(float2));
  float*  grad = (float*) alloc((size_t)BD*4*NPIX*sizeof(float));
  float*  pred = (float*) alloc((size_t)BD*4*NPIX*sizeof(float));
  float*  h1   = (float*) alloc((size_t)BD*FD*NPIX*sizeof(float));
  float*  h2   = (float*) alloc((size_t)BD*FD*NPIX*sizeof(float));
  float*  t1   = (float*) alloc((size_t)BD*FD*NPIX*sizeof(float));
  float*  t2   = (float*) alloc((size_t)BD*FD*NPIX*sizeof(float));
  h16*    wh   = (h16*)   alloc((size_t)109568*sizeof(h16));

  const h16* W1h  = wh;
  const h16* WihA = wh + 14336;
  const h16* WhhA = wh + 26624;
  const h16* WihB = wh + 38912;
  const h16* WhhB = wh + 51200;
  const h16* W2h  = wh + 63488;
  const h16* Wfh  = wh + 100352;

  k_convert_weights<<<(109568 + 255)/256, 256, 0, stream>>>(W1, Wih1, Whh1, W2, Wih2, Whh2, Wf, wh);
  k_init<<<(BD*FD*NPIX + 255)/256, 256, 0, stream>>>(h1, h2, pred, R2i, S0i, B0i, phii);
  k_sigmodel<<<(BD*ED*NPIX + 255)/256, 256, 0, stream>>>(R2i, S0i, B0i, phii, TEs, gamma, ps, base);
  k_fft_row_fwd<<<BD*ED*CDN*XD, 128, 0, stream>>>(ps, sens, buf);
  k_fft_col_resid<<<BD*ED*CDN*YD, 128, 0, stream>>>(buf, ksp, mask);
  k_ifft_row_cc<<<BD*ED*XD, 128, 0, stream>>>(buf, sens, cc);
  k_grad<<<(BD*NPIX + 255)/256, 256, 0, stream>>>(cc, ps, base, TEs, grad);

  int conv_grid = BD * XD * (YD/16);   // 8192 blocks, each 16 pixels along Y
  for (int step = 0; step < 8; ++step) {
    k_conv5<<<conv_grid, 128, 0, stream>>>(grad, pred, t1, W1h, b1);
    k_gru<<<conv_grid, 128, 0, stream>>>(t1, h1, WihA, WhhA, bih1, bhh1);
    k_conv3<<<conv_grid, 128, 0, stream>>>(h1, t2, W2h, b2, 2);
    k_gru<<<conv_grid, 128, 0, stream>>>(t2, h2, WihB, WhhB, bih2, bhh2);
    k_convf_update<<<conv_grid, 32, 0, stream>>>(h2, Wfh, pred, out + (size_t)step*BD*4*NPIX);
  }
}
